// Recommender_1340029796577
// MI455X (gfx1250) — matrix operand order repaired and checked by reading, not compiled, not run on hardware
//
#include <hip/hip_runtime.h>
#include <hip/hip_bf16.h>
#include <math.h>

#define N_USERS    100000
#define N_ITEMS    50000
#define N_ENTITIES 200000
#define DIM        64
#define KG_NNZ     2000000
#define II_NNZ     2000000
#define UI_NNZ     2000000
#define BATCH      4096
#define DECAY      1.0e-4f
#define ALPHA      0.2f

typedef _Float16 h8  __attribute__((ext_vector_type(8)));
typedef _Float16 h16 __attribute__((ext_vector_type(16)));
typedef float    f8  __attribute__((ext_vector_type(8)));

__device__ __forceinline__ void atomic_add_f32(float* p, float v) {
  unsafeAtomicAdd(p, v);   // non-returning global_atomic_add_f32
}

__device__ __forceinline__ float wave_sum32(float v) {
#pragma unroll
  for (int m = 16; m >= 1; m >>= 1) v += __shfl_xor(v, m, 32);
  return v;
}

// ---------------- KG SpMM, fused gather: out[row] += val * e[ent]*W[rel] ----
__global__ __launch_bounds__(256) void kg_spmm_kernel(
    const int* __restrict__ rows, const int* __restrict__ cols,
    const float* __restrict__ vals,
    const int* __restrict__ pair_ent, const int* __restrict__ pair_rel,
    const float* __restrict__ weight,
    const float* __restrict__ e_in, float* __restrict__ e_out) {
  long long tid = (long long)blockIdx.x * blockDim.x + threadIdx.x;
  int n = (int)(tid >> 4);
  if (n >= KG_NNZ) return;
  int c4 = ((int)tid & 15) * 4;
  int r = rows[n]; int c = cols[n]; float v = vals[n];
  int ent = pair_ent[c]; int rel = pair_rel[c] - 1;
  float4 ev = *(const float4*)(e_in + (size_t)ent * DIM + c4);
  float4 wv = *(const float4*)(weight + (size_t)rel * DIM + c4);
  float* o = e_out + (size_t)r * DIM + c4;
  atomic_add_f32(o + 0, v * ev.x * wv.x);
  atomic_add_f32(o + 1, v * ev.y * wv.y);
  atomic_add_f32(o + 2, v * ev.z * wv.z);
  atomic_add_f32(o + 3, v * ev.w * wv.w);
}

// ---------------- plain SpMM: out[row] += val * x[col] ----------------------
__global__ __launch_bounds__(256) void spmm_kernel(
    const int* __restrict__ rows, const int* __restrict__ cols,
    const float* __restrict__ vals,
    const float* __restrict__ x, float* __restrict__ out, int nnz) {
  long long tid = (long long)blockIdx.x * blockDim.x + threadIdx.x;
  int n = (int)(tid >> 4);
  if (n >= nnz) return;
  int c4 = ((int)tid & 15) * 4;
  int r = rows[n]; int c = cols[n]; float v = vals[n];
  float4 xv = *(const float4*)(x + (size_t)c * DIM + c4);
  float* o = out + (size_t)r * DIM + c4;
  atomic_add_f32(o + 0, v * xv.x);
  atomic_add_f32(o + 1, v * xv.y);
  atomic_add_f32(o + 2, v * xv.z);
  atomic_add_f32(o + 3, v * xv.w);
}

// ---------------- row-wise l2norm in place (x / max(||x||,1e-12)) -----------
__global__ __launch_bounds__(256) void l2norm_kernel(float* __restrict__ x, int nrows) {
  int row = (int)(((long long)blockIdx.x * blockDim.x + threadIdx.x) >> 5);
  int lane = threadIdx.x & 31;
  if (row >= nrows) return;
  float* p = x + (size_t)row * DIM + lane * 2;
  float2 v = *(float2*)p;
  float ss = wave_sum32(v.x * v.x + v.y * v.y);
  float inv = 1.0f / fmaxf(sqrtf(ss), 1e-12f);
  v.x *= inv; v.y *= inv;
  *(float2*)p = v;
}

// ---------------- acc += coeff * l2norm_rows(src) ---------------------------
__global__ __launch_bounds__(256) void l2norm_axpy_kernel(
    const float* __restrict__ src, float* __restrict__ acc, int nrows, float coeff) {
  int row = (int)(((long long)blockIdx.x * blockDim.x + threadIdx.x) >> 5);
  int lane = threadIdx.x & 31;
  if (row >= nrows) return;
  const float* p = src + (size_t)row * DIM + lane * 2;
  float2 v = *(const float2*)p;
  float ss = wave_sum32(v.x * v.x + v.y * v.y);
  float s = coeff / fmaxf(sqrtf(ss), 1e-12f);
  float* a = acc + (size_t)row * DIM + lane * 2;
  float2 av = *(float2*)a;
  av.x += s * v.x; av.y += s * v.y;
  *(float2*)a = av;
}

// ---------------- y += a*x --------------------------------------------------
__global__ __launch_bounds__(256) void axpy_kernel(
    const float* __restrict__ x, float* __restrict__ y, float a, int n) {
  int i = blockIdx.x * blockDim.x + threadIdx.x;
  if (i < n) y[i] += a * x[i];
}

// ---------------- batch gather + normalize + BPR/reg partials ---------------
__global__ __launch_bounds__(256) void batch_prep_kernel(
    const float* __restrict__ user_kg, const float* __restrict__ user_ii,
    const float* __restrict__ item_gcn,
    const int* __restrict__ users, const int* __restrict__ pos_items,
    const int* __restrict__ neg_items,
    _Float16* __restrict__ a16, _Float16* __restrict__ b16,
    float* __restrict__ diag, float* __restrict__ mf_part,
    float* __restrict__ reg_part) {
  int i = (int)(((long long)blockIdx.x * blockDim.x + threadIdx.x) >> 5);
  int lane = threadIdx.x & 31;
  if (i >= BATCH) return;
  int u = users[i], p = pos_items[i], q = neg_items[i];
  float2 kg = *(const float2*)(user_kg + (size_t)u * DIM + lane * 2);
  float2 ii = *(const float2*)(user_ii + (size_t)u * DIM + lane * 2);
  float sskg = wave_sum32(kg.x * kg.x + kg.y * kg.y);
  float ssii = wave_sum32(ii.x * ii.x + ii.y * ii.y);
  float ikg = 1.0f / sqrtf(sskg + 1e-24f);
  float iii = 1.0f / sqrtf(ssii + 1e-24f);
  float ax = kg.x * ikg, ay = kg.y * ikg;
  float bx = ii.x * iii, by = ii.y * iii;
  a16[(size_t)i * DIM + lane * 2 + 0] = (_Float16)ax;
  a16[(size_t)i * DIM + lane * 2 + 1] = (_Float16)ay;
  b16[(size_t)i * DIM + lane * 2 + 0] = (_Float16)bx;
  b16[(size_t)i * DIM + lane * 2 + 1] = (_Float16)by;
  float dg = wave_sum32(ax * bx + ay * by);
  float2 pe = *(const float2*)(item_gcn + (size_t)p * DIM + lane * 2);
  float2 ne = *(const float2*)(item_gcn + (size_t)q * DIM + lane * 2);
  float uex = kg.x + ii.x, uey = kg.y + ii.y;   // user_gcn = user_kg + user_ii
  float ps = wave_sum32(uex * pe.x + uey * pe.y);
  float ns = wave_sum32(uex * ne.x + uey * ne.y);
  float rg = wave_sum32(uex * uex + uey * uey + pe.x * pe.x + pe.y * pe.y +
                        ne.x * ne.x + ne.y * ne.y);
  if (lane == 0) {
    diag[i] = dg;
    float xx = ps - ns;
    float logsig = fminf(xx, 0.0f) - log1pf(__expf(-fabsf(xx)));
    mf_part[i] = -logsig;
    reg_part[i] = rg;
  }
}

// ---------------- fused SSL GEMM (WMMA f16) + softmax row-sums --------------
// scores[i,j] = a[i,:] . b[j,:], K=64 -> two chained v_wmma_f32_16x16x32_f16.
// scores are in [-1,1] (l2-normalized), so sum(exp) needs no max subtraction.
__global__ __launch_bounds__(256) void ssl_gemm_kernel(
    const _Float16* __restrict__ a16, const _Float16* __restrict__ b16,
    float* __restrict__ rowsum) {
  const int lane = threadIdx.x & 31;
  const int wave = threadIdx.x >> 5;
  const int r0 = (blockIdx.x * 8 + wave) * 16;   // 16-row strip per wave
  const int half = lane & 15;
  const int hi = lane >> 4;

  // A (16x32 f16) layout: row = r0 + (lane&15);
  // halves 0..15 = K kbase+hi*16 .. +15 contiguous (matches B)
  const _Float16* arow = a16 + (size_t)(r0 + half) * DIM;
  h8 lo0 = *(const h8*)(arow + 0 + hi * 16);
  h8 up0 = *(const h8*)(arow + 8 + hi * 16);
  h8 lo1 = *(const h8*)(arow + 32 + hi * 16);
  h8 up1 = *(const h8*)(arow + 40 + hi * 16);
  h16 A0, A1;
#pragma unroll
  for (int t = 0; t < 8; ++t) {
    A0[t] = lo0[t]; A0[t + 8] = up0[t];
    A1[t] = lo1[t]; A1[t + 8] = up1[t];
  }

  float sums[8];
#pragma unroll
  for (int v = 0; v < 8; ++v) sums[v] = 0.0f;

  for (int c0 = 0; c0 < BATCH; c0 += 16) {
    // B (32x16 f16): col = c0 + (lane&15); K = kbase + hi*16 .. +15 contiguous
    const _Float16* bcol = b16 + (size_t)(c0 + half) * DIM + hi * 16;
    h16 B0 = *(const h16*)(bcol);
    h16 B1 = *(const h16*)(bcol + 32);
    f8 c = {};
    c = __builtin_amdgcn_wmma_f32_16x16x32_f16(false, A0, false, B0,
                                               (short)0, c, false, false);
    c = __builtin_amdgcn_wmma_f32_16x16x32_f16(false, A1, false, B1,
                                               (short)0, c, false, false);
    // D layout: this lane's column is c0+half; VGPR v holds row r0 + v + hi*8
#pragma unroll
    for (int v = 0; v < 8; ++v) sums[v] += __expf(c[v]);
  }
  // reduce each row over the 16 columns held by this half-wave (xor <= 8 stays
  // inside the 16-lane group)
#pragma unroll
  for (int v = 0; v < 8; ++v) {
    float s = sums[v];
#pragma unroll
    for (int m = 8; m >= 1; m >>= 1) s += __shfl_xor(s, m, 32);
    sums[v] = s;
  }
  if (half == 0) {
#pragma unroll
    for (int v = 0; v < 8; ++v) rowsum[r0 + hi * 8 + v] = sums[v];
  }
}

// ---------------- final scalar reduction ------------------------------------
__global__ __launch_bounds__(256) void final_reduce_kernel(
    const float* __restrict__ mf_part, const float* __restrict__ reg_part,
    const float* __restrict__ diag, const float* __restrict__ rowsum,
    float* __restrict__ out) {
  __shared__ float smf[256], srg[256], sss[256];
  int t = threadIdx.x;
  float mf = 0.0f, rg = 0.0f, ssl = 0.0f;
  for (int i = t; i < BATCH; i += 256) {
    mf += mf_part[i];
    rg += reg_part[i];
    ssl += logf(rowsum[i]) - diag[i];   // logsumexp_i - scores[i,i]
  }
  smf[t] = mf; srg[t] = rg; sss[t] = ssl;
  __syncthreads();
  for (int s = 128; s > 0; s >>= 1) {
    if (t < s) { smf[t] += smf[t + s]; srg[t] += srg[t + s]; sss[t] += sss[t + s]; }
    __syncthreads();
  }
  if (t == 0) {
    float mfl = smf[0] / (float)BATCH;
    float embl = DECAY * (srg[0] * 0.5f) / (float)BATCH;
    float ssll = sss[0] / (float)BATCH;
    out[0] = mfl + embl + ALPHA * ssll;
    out[1] = mfl;
    out[2] = embl;
  }
}

extern "C" void kernel_launch(void* const* d_in, const int* in_sizes, int n_in,
                              void* d_out, int out_size, void* d_ws, size_t ws_size,
                              hipStream_t stream) {
  (void)in_sizes; (void)n_in; (void)out_size; (void)ws_size;
  const float* all_embed  = (const float*)d_in[0];
  const float* weight     = (const float*)d_in[1];
  const float* kg_vals    = (const float*)d_in[2];
  const float* ii_vals    = (const float*)d_in[3];
  const float* ui_vals    = (const float*)d_in[4];
  const int* kg_pair_rel  = (const int*)d_in[5];
  const int* kg_pair_ent  = (const int*)d_in[6];
  const int* kg_rows      = (const int*)d_in[7];
  const int* kg_cols      = (const int*)d_in[8];
  const int* ii_rows      = (const int*)d_in[9];
  const int* ii_cols      = (const int*)d_in[10];
  const int* ui_rows      = (const int*)d_in[11];
  const int* ui_cols      = (const int*)d_in[12];
  const int* users        = (const int*)d_in[13];
  const int* pos_items    = (const int*)d_in[14];
  const int* neg_items    = (const int*)d_in[15];
  float* out = (float*)d_out;

  char* ws = (char*)d_ws;
  size_t off = 0;
  auto alloc = [&](size_t bytes) -> char* {
    char* p = ws + off;
    off = (off + bytes + 255) & ~(size_t)255;
    return p;
  };
  const size_t EB = (size_t)N_ENTITIES * DIM * 4;
  const size_t IB = (size_t)N_ITEMS * DIM * 4;
  const size_t UB = (size_t)N_USERS * DIM * 4;
  float* e1        = (float*)alloc(EB);
  float* e2        = (float*)alloc(EB);
  float* item_agg  = (float*)alloc(IB);
  float* item_gcn  = (float*)alloc(IB);
  float* user_kg   = (float*)alloc(UB);
  float* user_ii   = (float*)alloc(UB);
  float* utmp      = (float*)alloc(UB);
  _Float16* a16    = (_Float16*)alloc((size_t)BATCH * DIM * 2);
  _Float16* b16    = (_Float16*)alloc((size_t)BATCH * DIM * 2);
  float* diag      = (float*)alloc((size_t)BATCH * 4);
  float* mf_part   = (float*)alloc((size_t)BATCH * 4);
  float* reg_part  = (float*)alloc((size_t)BATCH * 4);
  float* rowsum    = (float*)alloc((size_t)BATCH * 4);

  const float* entity0 = all_embed + (size_t)N_USERS * DIM;  // entity_emb0 (item_emb0 = first N_ITEMS rows)

  const int spmm_blk   = (int)(((long long)KG_NNZ * 16 + 255) / 256);  // also II/UI (same nnz)
  const int ln_ent_blk = (N_ENTITIES * 32 + 255) / 256;
  const int ln_itm_blk = (N_ITEMS * 32 + 255) / 256;
  const int ln_usr_blk = (N_USERS * 32 + 255) / 256;
  const int ax_itm_blk = (N_ITEMS * DIM + 255) / 256;

  // ---- init accumulators ----
  hipMemcpyAsync(user_kg, all_embed, UB, hipMemcpyDeviceToDevice, stream);
  hipMemcpyAsync(user_ii, all_embed, UB, hipMemcpyDeviceToDevice, stream);
  hipMemsetAsync(item_gcn, 0, IB, stream);
  // item_gcn starts at 2*item_emb0 (item_kg_multi[0] == item_multi[0] == item_emb0)
  axpy_kernel<<<ax_itm_blk, 256, 0, stream>>>(entity0, item_gcn, 2.0f, N_ITEMS * DIM);

  // ---- item_agg = l2norm(spmm(ii, item_emb0)); identical for all 3 hops ----
  hipMemsetAsync(item_agg, 0, IB, stream);
  spmm_kernel<<<spmm_blk, 256, 0, stream>>>(ii_rows, ii_cols, ii_vals, entity0, item_agg, II_NNZ);
  l2norm_kernel<<<ln_itm_blk, 256, 0, stream>>>(item_agg, N_ITEMS);
  axpy_kernel<<<ax_itm_blk, 256, 0, stream>>>(item_agg, item_gcn, 3.0f, N_ITEMS * DIM);

  // ---- ui SpMM on item_emb0 (shared between user_kg and user_ii) ----
  hipMemsetAsync(utmp, 0, UB, stream);
  spmm_kernel<<<spmm_blk, 256, 0, stream>>>(ui_rows, ui_cols, ui_vals, entity0, utmp, UI_NNZ);
  l2norm_axpy_kernel<<<ln_usr_blk, 256, 0, stream>>>(utmp, user_kg, N_USERS, 1.0f);
  l2norm_axpy_kernel<<<ln_usr_blk, 256, 0, stream>>>(utmp, user_ii, N_USERS, 1.0f);

  // ---- ui SpMM on item_agg (used twice in user_ii) ----
  hipMemsetAsync(utmp, 0, UB, stream);
  spmm_kernel<<<spmm_blk, 256, 0, stream>>>(ui_rows, ui_cols, ui_vals, item_agg, utmp, UI_NNZ);
  l2norm_axpy_kernel<<<ln_usr_blk, 256, 0, stream>>>(utmp, user_ii, N_USERS, 2.0f);

  // ---- hop 1: e1 = l2norm(KG-spmm(entity0)) ----
  hipMemsetAsync(e1, 0, EB, stream);
  kg_spmm_kernel<<<spmm_blk, 256, 0, stream>>>(kg_rows, kg_cols, kg_vals,
      kg_pair_ent, kg_pair_rel, weight, entity0, e1);
  l2norm_kernel<<<ln_ent_blk, 256, 0, stream>>>(e1, N_ENTITIES);
  axpy_kernel<<<ax_itm_blk, 256, 0, stream>>>(e1, item_gcn, 1.0f, N_ITEMS * DIM);
  hipMemsetAsync(utmp, 0, UB, stream);
  spmm_kernel<<<spmm_blk, 256, 0, stream>>>(ui_rows, ui_cols, ui_vals, e1, utmp, UI_NNZ);
  l2norm_axpy_kernel<<<ln_usr_blk, 256, 0, stream>>>(utmp, user_kg, N_USERS, 1.0f);

  // ---- hop 2: e2 = l2norm(KG-spmm(e1)) ----
  hipMemsetAsync(e2, 0, EB, stream);
  kg_spmm_kernel<<<spmm_blk, 256, 0, stream>>>(kg_rows, kg_cols, kg_vals,
      kg_pair_ent, kg_pair_rel, weight, e1, e2);
  l2norm_kernel<<<ln_ent_blk, 256, 0, stream>>>(e2, N_ENTITIES);
  axpy_kernel<<<ax_itm_blk, 256, 0, stream>>>(e2, item_gcn, 1.0f, N_ITEMS * DIM);
  hipMemsetAsync(utmp, 0, UB, stream);
  spmm_kernel<<<spmm_blk, 256, 0, stream>>>(ui_rows, ui_cols, ui_vals, e2, utmp, UI_NNZ);
  l2norm_axpy_kernel<<<ln_usr_blk, 256, 0, stream>>>(utmp, user_kg, N_USERS, 1.0f);

  // ---- hop 3: reuse e1 (its slice already consumed) ----
  hipMemsetAsync(e1, 0, EB, stream);
  kg_spmm_kernel<<<spmm_blk, 256, 0, stream>>>(kg_rows, kg_cols, kg_vals,
      kg_pair_ent, kg_pair_rel, weight, e2, e1);
  l2norm_kernel<<<ln_ent_blk, 256, 0, stream>>>(e1, N_ENTITIES);
  axpy_kernel<<<ax_itm_blk, 256, 0, stream>>>(e1, item_gcn, 1.0f, N_ITEMS * DIM);

  // ---- batch stage: gather, normalize, BPR/reg partials, f16 staging ----
  batch_prep_kernel<<<(BATCH * 32) / 256, 256, 0, stream>>>(
      user_kg, user_ii, item_gcn, users, pos_items, neg_items,
      a16, b16, diag, mf_part, reg_part);

  // ---- SSL GEMM via WMMA: 256 waves x 16-row strips, full 4096 cols ----
  ssl_gemm_kernel<<<32, 256, 0, stream>>>(a16, b16, rowsum);

  // ---- final losses ----
  final_reduce_kernel<<<1, 256, 0, stream>>>(mf_part, reg_part, diag, rowsum, out);
}
